// EGC_9131100472074
// MI455X (gfx1250) — matrix operand order repaired and checked by reading, not compiled, not run on hardware
//
#include <hip/hip_runtime.h>
#include <hip/hip_bf16.h>
#include <math.h>

// ---------------------------------------------------------------------------
// EGConv-style GNN forward for MI455X (gfx1250, wave32).
// Dense node transforms: V_WMMA_F32_16X16X32_BF16, f32 accumulate.
//   - packed weights staged global->LDS once per block with
//     GLOBAL_LOAD_ASYNC_TO_LDS_B128 (ASYNCcnt + barrier), B fragments then
//     come from LDS (ds_load_b128) -> WMMA issue not gated on loadcnt
//   - A fragments loaded as 4x global_load_b128 (two contiguous 8-float runs per CDNA5 layout)
//   - one wave computes 16x64 output (4 tiles) reusing the A fragment
// Edge aggregation: L2-resident float atomics (agg arrays ~38MB << 192MB L2).
// ---------------------------------------------------------------------------

#define NGRAPH 100
#define CDIM   128       // hidden channels
#define HD     8         // heads
#define BBASE  4         // bases
#define FH     16        // CDIM / HD
#define BD     64        // BBASE * FH
#define LLAY   4         // EGConv layers

typedef __attribute__((ext_vector_type(16))) __bf16 v16bf;
typedef __attribute__((ext_vector_type(8)))  float  v8f;

// ------------------------------ utilities ---------------------------------

__global__ void k_fill(float* __restrict__ p, int n, float v) {
  int i = blockIdx.x * blockDim.x + threadIdx.x;
  if (i < n) p[i] = v;
}

__device__ __forceinline__ void atomicMaxFloat(float* addr, float v) {
  // ordered-int trick -> pure GLOBAL_ATOMIC_MAX/MIN, no CAS loop
  if (v >= 0.0f) atomicMax((int*)addr, __float_as_int(v));
  else           atomicMin((unsigned int*)addr, __float_as_uint(v));
}

// ------------------------------ degrees -----------------------------------

__global__ void k_deg_edges(const long long* __restrict__ ei, int E,
                            float* __restrict__ deg) {
  int e = blockIdx.x * blockDim.x + threadIdx.x;
  if (e < E) atomicAdd(&deg[(int)ei[E + e]], 1.0f);   // dst row
}

__global__ void k_dinv(const float* __restrict__ deg, float* __restrict__ dinv,
                       int N) {
  int i = blockIdx.x * blockDim.x + threadIdx.x;
  if (i < N) dinv[i] = rsqrtf(deg[i]);   // deg >= 1 (self-loop)
}

// ------------------------------ GCN layer ---------------------------------

__global__ void k_gcn_node(const float* __restrict__ x,
                           const float* __restrict__ gw,
                           const float* __restrict__ gb,
                           const float* __restrict__ dinv,
                           float* __restrict__ h, float* __restrict__ xcur,
                           int N) {
  int idx = blockIdx.x * blockDim.x + threadIdx.x;
  if (idx >= N * CDIM) return;
  int n = idx >> 7, c = idx & 127;
  float hv = x[n * 2] * gw[c] + x[n * 2 + 1] * gw[CDIM + c];
  h[idx] = hv;
  float di = dinv[n];
  xcur[idx] = gb[c] + hv * di * di;
}

// one wave per edge, float4 per lane covers 128 channels
__global__ void k_gcn_edges(const long long* __restrict__ ei, int E,
                            const float* __restrict__ dinv,
                            const float* __restrict__ h,
                            float* __restrict__ xcur) {
  int wid  = (blockIdx.x * blockDim.x + threadIdx.x) >> 5;
  int lane = threadIdx.x & 31;
  if (wid >= E) return;
  int s = (int)ei[wid], d = (int)ei[E + wid];
  float coef = dinv[s] * dinv[d];
  float4 v = ((const float4*)(h + (size_t)s * CDIM))[lane];
  float* o = xcur + (size_t)d * CDIM + lane * 4;
  atomicAdd(o + 0, v.x * coef);
  atomicAdd(o + 1, v.y * coef);
  atomicAdd(o + 2, v.z * coef);
  atomicAdd(o + 3, v.w * coef);
}

// ------------------- weight pre-pack into WMMA fragment order --------------
// PW layout: [tcol][kc][lane] -> 16 bf16 (32B) per lane, lanes contiguous.

__global__ void k_pack_w(const float* __restrict__ W,
                         unsigned int* __restrict__ PW, int K, int NC) {
  int idx = blockIdx.x * blockDim.x + threadIdx.x;
  int KC = K >> 5;
  int total = (NC >> 4) * KC * 32;
  if (idx >= total) return;
  int lane = idx & 31;
  int kc   = (idx >> 5) % KC;
  int tcol = idx / (32 * KC);
  int n = lane & 15, half = lane >> 4;
  unsigned int* o = PW + (size_t)idx * 8;
#pragma unroll
  for (int v = 0; v < 8; ++v) {
    int k = (kc << 5) + 8 * half + 2 * v + (v >= 4 ? 8 : 0);
    __bf16 lo = (__bf16)W[(size_t)k * NC + (tcol << 4) + n];
    __bf16 hi = (__bf16)W[(size_t)(k + 1) * NC + (tcol << 4) + n];
    unsigned int ulo = (unsigned int)__builtin_bit_cast(unsigned short, lo);
    unsigned int uhi = (unsigned int)__builtin_bit_cast(unsigned short, hi);
    o[v] = ulo | (uhi << 16);
  }
}

// ------------------- WMMA GEMM  Y = X[N,K] @ W[K,NC] (+bias) ---------------
// Packed weights async-copied to LDS once per block; one wave computes a
// 16x64 output strip (4 WMMA tiles), reusing the A fragment.

__global__ void k_wmma_gemm4(const float* __restrict__ X,
                             const unsigned int* __restrict__ PW,
                             const float* __restrict__ bias,
                             float* __restrict__ Y, int N, int K, int NC) {
  __shared__ unsigned int lpw[8192];            // up to 32KB packed weights
  const int KC = K >> 5;
  const int nbytes = (NC >> 4) * KC * 32 * 32;  // total packed bytes

  // cooperative async copy global->LDS (uniform trip count: nbytes % 4096==0)
  for (int off = threadIdx.x * 16; off < nbytes; off += blockDim.x * 16) {
    unsigned int lds_addr =
        (unsigned int)(size_t)(&lpw[0]) + (unsigned int)off;  // addr[31:0]
    unsigned long long gaddr = (unsigned long long)((const char*)PW + off);
    asm volatile("global_load_async_to_lds_b128 %0, %1, off"
                 :: "v"(lds_addr), "v"(gaddr)
                 : "memory");
  }
  asm volatile("s_wait_asynccnt 0x0" ::: "memory");
  __syncthreads();

  int lane   = threadIdx.x & 31;
  int wid    = (blockIdx.x * blockDim.x + threadIdx.x) >> 5;
  int groups = NC >> 6;                       // 64-wide column groups
  int total  = (N >> 4) * groups;
  if (wid >= total) return;                   // uniform per wave -> EXEC all-1s
  int trow = wid / groups, g = wid - trow * groups;
  int row0 = trow << 4, col0 = g << 6;
  int m = lane & 15, half = lane >> 4;

  const v16bf* lb = (const v16bf*)lpw;        // 32B fragments in LDS
  v8f acc[4] = {{}, {}, {}, {}};
  for (int kc = 0; kc < KC; ++kc) {
    int k0 = kc << 5;
    // A fragment: two contiguous 8-float runs per lane -> 4x b128 loads
    const float* xr = X + (size_t)(row0 + m) * K + k0 + 8 * half;
    float4 q0 = ((const float4*)xr)[0];
    float4 q1 = ((const float4*)xr)[1];
    float4 q2 = ((const float4*)(xr + 16))[0];
    float4 q3 = ((const float4*)(xr + 16))[1];
    v16bf a;
    a[0]  = (__bf16)q0.x; a[1]  = (__bf16)q0.y;
    a[2]  = (__bf16)q0.z; a[3]  = (__bf16)q0.w;
    a[4]  = (__bf16)q1.x; a[5]  = (__bf16)q1.y;
    a[6]  = (__bf16)q1.z; a[7]  = (__bf16)q1.w;
    a[8]  = (__bf16)q2.x; a[9]  = (__bf16)q2.y;
    a[10] = (__bf16)q2.z; a[11] = (__bf16)q2.w;
    a[12] = (__bf16)q3.x; a[13] = (__bf16)q3.y;
    a[14] = (__bf16)q3.z; a[15] = (__bf16)q3.w;
    // preload all 4 B fragments from LDS, then issue 4 WMMAs back-to-back
    v16bf b0 = lb[((g * 4 + 0) * KC + kc) * 32 + lane];
    v16bf b1 = lb[((g * 4 + 1) * KC + kc) * 32 + lane];
    v16bf b2 = lb[((g * 4 + 2) * KC + kc) * 32 + lane];
    v16bf b3 = lb[((g * 4 + 3) * KC + kc) * 32 + lane];
    acc[0] = __builtin_amdgcn_wmma_f32_16x16x32_bf16(
        false, a, false, b0, (short)0, acc[0], false, false);
    acc[1] = __builtin_amdgcn_wmma_f32_16x16x32_bf16(
        false, a, false, b1, (short)0, acc[1], false, false);
    acc[2] = __builtin_amdgcn_wmma_f32_16x16x32_bf16(
        false, a, false, b2, (short)0, acc[2], false, false);
    acc[3] = __builtin_amdgcn_wmma_f32_16x16x32_bf16(
        false, a, false, b3, (short)0, acc[3], false, false);
  }
#pragma unroll
  for (int t = 0; t < 4; ++t) {
    int cb = col0 + t * 16 + m;
    float bv = bias ? bias[cb] : 0.0f;
#pragma unroll
    for (int r = 0; r < 8; ++r)
      Y[(size_t)(row0 + r + 8 * half) * NC + cb] = acc[t][r] + bv;
  }
}

// ------------------------ edge aggregation (per layer) ---------------------

__global__ void k_agg_init(const float* __restrict__ bases,
                           float* __restrict__ asum, float* __restrict__ asq,
                           float* __restrict__ amax, int N) {
  int idx = blockIdx.x * blockDim.x + threadIdx.x;
  if (idx >= N * BD) return;
  float v = bases[idx];
  asum[idx] = v;
  asq[idx]  = v * v;
  amax[idx] = v;
}

__global__ void k_agg_edges(const long long* __restrict__ ei, int E,
                            const float* __restrict__ bases,
                            float* __restrict__ asum, float* __restrict__ asq,
                            float* __restrict__ amax) {
  int wid  = (blockIdx.x * blockDim.x + threadIdx.x) >> 5;
  int lane = threadIdx.x & 31;
  if (wid >= E) return;
  int s = (int)ei[wid], d = (int)ei[E + wid];
  float2 v = ((const float2*)(bases + (size_t)s * BD))[lane];
  int o = d * BD + lane * 2;
  atomicAdd(&asum[o],     v.x);
  atomicAdd(&asum[o + 1], v.y);
  atomicAdd(&asq[o],      v.x * v.x);
  atomicAdd(&asq[o + 1],  v.y * v.y);
  atomicMaxFloat(&amax[o],     v.x);
  atomicMaxFloat(&amax[o + 1], v.y);
}

__global__ void k_combine(const float* __restrict__ wco,
                          const float* __restrict__ asum,
                          const float* __restrict__ asq,
                          const float* __restrict__ amax,
                          const float* __restrict__ deg,
                          const float* __restrict__ convb,
                          float* __restrict__ h, int N) {
  int idx = blockIdx.x * blockDim.x + threadIdx.x;
  if (idx >= N * CDIM) return;
  int n = idx >> 7, c = idx & 127;
  int hh = c >> 4, f = c & 15;
  float inv = 1.0f / deg[n];
  const float* wp = wco + (size_t)n * CDIM + hh * 16;
  float acc = 0.0f;
#pragma unroll
  for (int b = 0; b < BBASE; ++b) {
    int o = n * BD + b * 16 + f;
    float s = asum[o], q = asq[o], mx = amax[o];
    float mean = s * inv;
    float sd = sqrtf(fmaxf(q * inv - mean * mean, 0.0f) + 1e-5f);
    acc += wp[b] * s + wp[4 + b] * mean + wp[8 + b] * mx + wp[12 + b] * sd;
  }
  h[idx] = acc + convb[c];
}

// ------------------------ batchnorm (training-mode stats) ------------------

__global__ void k_colstats(const float* __restrict__ H, int n, int C,
                           float* __restrict__ stats) {
  int c = threadIdx.x;
  float s = 0.0f, q = 0.0f;
  for (int r = blockIdx.x; r < n; r += gridDim.x) {
    float v = H[(size_t)r * C + c];
    s += v;
    q += v * v;
  }
  atomicAdd(&stats[c], s);
  atomicAdd(&stats[C + c], q);
}

__global__ void k_bn_act(const float* __restrict__ H,
                         const float* __restrict__ stats,
                         const float* __restrict__ g,
                         const float* __restrict__ b, float* __restrict__ xio,
                         int n, int C, float invN, int residual) {
  int idx = blockIdx.x * blockDim.x + threadIdx.x;
  if (idx >= n * C) return;
  int c = idx % C;
  float m   = stats[c] * invN;
  float var = stats[C + c] * invN - m * m;
  float y = (H[idx] - m) * rsqrtf(fmaxf(var, 0.0f) + 1e-5f) * g[c] + b[c];
  y = fmaxf(y, 0.0f);
  if (residual) xio[idx] += y;
  else          xio[idx] = y;
}

// ------------------------ pooling + MLP head -------------------------------

__global__ void k_pool(const float* __restrict__ x,
                       const long long* __restrict__ batch,
                       float* __restrict__ gsum, float* __restrict__ gcnt,
                       int N) {
  int idx = blockIdx.x * blockDim.x + threadIdx.x;
  if (idx >= N * CDIM) return;
  int n = idx >> 7, c = idx & 127;
  int gi = (int)batch[n];
  atomicAdd(&gsum[gi * CDIM + c], x[idx]);
  if (c == 0) atomicAdd(&gcnt[gi], 1.0f);
}

__global__ void k_gdiv(float* __restrict__ gsum, const float* __restrict__ gcnt) {
  int idx = blockIdx.x * blockDim.x + threadIdx.x;
  if (idx >= NGRAPH * CDIM) return;
  gsum[idx] /= fmaxf(gcnt[idx >> 7], 1.0f);
}

__global__ void k_gemm_small(const float* __restrict__ A,
                             const float* __restrict__ W,
                             float* __restrict__ Y, int n, int K, int NC) {
  int idx = blockIdx.x * blockDim.x + threadIdx.x;
  if (idx >= n * NC) return;
  int i = idx / NC, j = idx - i * NC;
  float acc = 0.0f;
  for (int k = 0; k < K; ++k) acc += A[(size_t)i * K + k] * W[(size_t)k * NC + j];
  Y[idx] = acc;
}

__global__ void k_mlp3(const float* __restrict__ G2,
                       const float* __restrict__ w3,
                       const float* __restrict__ b3, float* __restrict__ out) {
  int g = blockIdx.x * blockDim.x + threadIdx.x;
  if (g >= NGRAPH) return;
  float acc = b3[0];
#pragma unroll
  for (int k = 0; k < 32; ++k) acc += G2[g * 32 + k] * w3[k];
  out[g] = acc;
}

// ---------------------------------------------------------------------------

extern "C" void kernel_launch(void* const* d_in, const int* in_sizes, int n_in,
                              void* d_out, int out_size, void* d_ws,
                              size_t ws_size, hipStream_t stream) {
  const float*     x       = (const float*)d_in[0];
  const long long* ei      = (const long long*)d_in[1];   // int64 [2,E]
  const long long* batch   = (const long long*)d_in[2];   // int64 [N]
  const float*     gcn_w   = (const float*)d_in[3];
  const float*     gcn_b   = (const float*)d_in[4];
  const float*     bases_w = (const float*)d_in[5];       // [L,128,64]
  const float*     comb_w  = (const float*)d_in[6];       // [L,128,128]
  const float*     comb_b  = (const float*)d_in[7];       // [L,128]
  const float*     conv_b  = (const float*)d_in[8];       // [L,128]
  const float*     bn_g    = (const float*)d_in[9];
  const float*     bn_b    = (const float*)d_in[10];
  const float*     mlp_w1  = (const float*)d_in[11];
  const float*     bn1_g   = (const float*)d_in[12];
  const float*     bn1_b   = (const float*)d_in[13];
  const float*     mlp_w2  = (const float*)d_in[14];
  const float*     bn2_g   = (const float*)d_in[15];
  const float*     bn2_b   = (const float*)d_in[16];
  const float*     mlp_w3  = (const float*)d_in[17];
  const float*     mlp_b3  = (const float*)d_in[18];
  float*           out     = (float*)d_out;

  const int N = in_sizes[0] / 2;   // 50000 (multiple of 16)
  const int E = in_sizes[1] / 2;   // 800000

  // workspace carve-up (floats), blocks padded to 64-float multiples so the
  // packed-weight region stays 32B aligned for b128 access
  auto pad64 = [](size_t v) { return (v + 63) & ~(size_t)63; };
  float* p = (float*)d_ws;
  float* deg   = p; p += pad64(N);
  float* dinv  = p; p += pad64(N);
  float* xcur  = p; p += pad64((size_t)N * CDIM);
  float* tmp   = p; p += pad64((size_t)N * CDIM);
  float* bases = p; p += pad64((size_t)N * BD);
  float* wco   = p; p += pad64((size_t)N * CDIM);
  float* asum  = p; p += pad64((size_t)N * BD);
  float* asq   = p; p += pad64((size_t)N * BD);
  float* amax  = p; p += pad64((size_t)N * BD);
  float* stats = p; p += pad64(2 * CDIM);
  float* gsum  = p; p += pad64(NGRAPH * CDIM);
  float* gcnt  = p; p += pad64(NGRAPH);
  float* g1    = p; p += pad64(NGRAPH * (CDIM / 2));
  float* g2    = p; p += pad64(NGRAPH * (CDIM / 4));
  unsigned int* pwb = (unsigned int*)p; p += pad64((BD / 16) * 4 * 32 * 8);
  unsigned int* pwc = (unsigned int*)p; p += pad64((CDIM / 16) * 4 * 32 * 8);

  const int T = 256;
  const int NC128 = N * CDIM, NC64 = N * BD;
  dim3 gNC((NC128 + T - 1) / T), gN64((NC64 + T - 1) / T);
  dim3 gE1((E + T - 1) / T);                  // thread/edge
  dim3 gEw((E * 32 + T - 1) / T);             // wave/edge
  dim3 gN((N + T - 1) / T);

  // ---- degrees ----
  k_fill<<<gN, T, 0, stream>>>(deg, N, 1.0f);              // self-loops
  k_deg_edges<<<gE1, T, 0, stream>>>(ei, E, deg);
  k_dinv<<<gN, T, 0, stream>>>(deg, dinv, N);

  // ---- GCNConv ----
  k_gcn_node<<<gNC, T, 0, stream>>>(x, gcn_w, gcn_b, dinv, tmp, xcur, N);
  k_gcn_edges<<<gEw, T, 0, stream>>>(ei, E, dinv, tmp, xcur);

  // ---- EGConv layers ----
  for (int l = 0; l < LLAY; ++l) {
    // pack this layer's weights into WMMA fragment order (bf16)
    k_pack_w<<<2, T, 0, stream>>>(bases_w + (size_t)l * CDIM * BD, pwb, CDIM,
                                  BD);
    k_pack_w<<<4, T, 0, stream>>>(comb_w + (size_t)l * CDIM * CDIM, pwc, CDIM,
                                  CDIM);

    int wb = (N / 16) * (BD / 64);      // waves for bases GEMM
    int wc = (N / 16) * (CDIM / 64);    // waves for comb GEMM
    k_wmma_gemm4<<<(wb * 32 + T - 1) / T, T, 0, stream>>>(
        xcur, pwb, nullptr, bases, N, CDIM, BD);
    k_wmma_gemm4<<<(wc * 32 + T - 1) / T, T, 0, stream>>>(
        xcur, pwc, comb_b + l * CDIM, wco, N, CDIM, CDIM);

    k_agg_init<<<gN64, T, 0, stream>>>(bases, asum, asq, amax, N);
    k_agg_edges<<<gEw, T, 0, stream>>>(ei, E, bases, asum, asq, amax);
    k_combine<<<gNC, T, 0, stream>>>(wco, asum, asq, amax, deg,
                                     conv_b + l * CDIM, tmp, N);

    k_fill<<<(2 * CDIM + T - 1) / T, T, 0, stream>>>(stats, 2 * CDIM, 0.0f);
    k_colstats<<<256, CDIM, 0, stream>>>(tmp, N, CDIM, stats);
    k_bn_act<<<gNC, T, 0, stream>>>(tmp, stats, bn_g + l * CDIM,
                                    bn_b + l * CDIM, xcur, N, CDIM,
                                    1.0f / (float)N, 1);
  }

  // ---- global mean pool ----
  k_fill<<<(NGRAPH * CDIM + T - 1) / T, T, 0, stream>>>(gsum, NGRAPH * CDIM,
                                                        0.0f);
  k_fill<<<1, T, 0, stream>>>(gcnt, NGRAPH, 0.0f);
  k_pool<<<gNC, T, 0, stream>>>(xcur, batch, gsum, gcnt, N);
  k_gdiv<<<(NGRAPH * CDIM + T - 1) / T, T, 0, stream>>>(gsum, gcnt);

  // ---- MLP head (100 rows; scalar GEMMs) ----
  k_gemm_small<<<(NGRAPH * 64 + T - 1) / T, T, 0, stream>>>(gsum, mlp_w1, g1,
                                                            NGRAPH, 128, 64);
  k_fill<<<1, 128, 0, stream>>>(stats, 128, 0.0f);
  k_colstats<<<4, 64, 0, stream>>>(g1, NGRAPH, 64, stats);
  k_bn_act<<<(NGRAPH * 64 + T - 1) / T, T, 0, stream>>>(
      g1, stats, bn1_g, bn1_b, g1, NGRAPH, 64, 1.0f / (float)NGRAPH, 0);

  k_gemm_small<<<(NGRAPH * 32 + T - 1) / T, T, 0, stream>>>(g1, mlp_w2, g2,
                                                            NGRAPH, 64, 32);
  k_fill<<<1, 64, 0, stream>>>(stats, 64, 0.0f);
  k_colstats<<<4, 32, 0, stream>>>(g2, NGRAPH, 32, stats);
  k_bn_act<<<(NGRAPH * 32 + T - 1) / T, T, 0, stream>>>(
      g2, stats, bn2_g, bn2_b, g2, NGRAPH, 32, 1.0f / (float)NGRAPH, 0);

  k_mlp3<<<1, NGRAPH, 0, stream>>>(g2, mlp_w3, mlp_b3, out);
}
